// MinGRU_17154099380272
// MI455X (gfx1250) — compile-verified, matrix operand order checked by reading, not run on hardware
//
#include <hip/hip_runtime.h>
#include <hip/hip_bf16.h>

typedef __attribute__((ext_vector_type(16))) __bf16 v16bf;
typedef __attribute__((ext_vector_type(8)))  float  v8f;

union FragU { uint4 u[2]; v16bf v; };

#define B_   8
#define S_   4096
#define D_   512
#define H_   512
#define M_   (B_*S_)      /* 32768 rows */
#define KDIM 512
#define NC   32           /* scan chunks */
#define CL   (S_/NC)      /* 128 steps per chunk */

// ---------------- helpers ----------------

__device__ __forceinline__ unsigned short f2bf(float x) {
  union { float f; unsigned int u; } v; v.f = x;
  unsigned int u = v.u;
  return (unsigned short)((u + 0x7FFFu + ((u >> 16) & 1u)) >> 16);  // RNE
}

__device__ __forceinline__ float softplusf_(float x) {
  return fmaxf(x, 0.0f) + log1pf(expf(-fabsf(x)));
}
__device__ __forceinline__ float log_g_(float x) {
  return (x >= 0.0f) ? logf(x + 0.5f) : -softplusf_(-x);
}
__device__ __forceinline__ float logaddexpf_(float p, float q) {
  float m = fmaxf(p, q);
  if (m == -INFINITY) return -INFINITY;
  return m + log1pf(expf(-fabsf(p - q)));
}

// ---------------- f32 -> bf16 convert ----------------

__global__ void k_f32_to_bf16(const float* __restrict__ src,
                              unsigned short* __restrict__ dst, int n) {
  int i = blockIdx.x * blockDim.x + threadIdx.x;
  if (i < n) dst[i] = f2bf(src[i]);
}

// ---------------- fused GEMM (bf16 WMMA) + MinGRU transform ----------------
// A fragment per ISA 16-bit 16x32 layout: lane = {r = lane&15 (row), hf = lane>>4}
// lanes 0-15: VGPR0-3 = K[k..k+7],  VGPR4-7 = K[k+16..k+23]
// lanes16-31: VGPR0-3 = K[k+8..k+15], VGPR4-7 = K[k+24..k+31]
__device__ __forceinline__ v16bf load_frag(const unsigned short* __restrict__ base,
                                           int row0, int k, int r, int hf) {
  const unsigned short* p = base + (size_t)(row0 + r) * KDIM + k + hf * 8;
  FragU f;
  f.u[0] = *(const uint4*)(p);
  f.u[1] = *(const uint4*)(p + 16);
  return f.v;
}

__global__ void __launch_bounds__(256)
k_gemm_fused(const unsigned short* __restrict__ X,   // [M_, 512] bf16, K contiguous
             const unsigned short* __restrict__ W,   // [1024, 512] bf16, K contiguous
             const float* __restrict__ bias,         // [1024]
             float* __restrict__ aOut,               // [M_, H_]  log_z + log_g(hidden)
             float* __restrict__ cOut) {             // [M_, H_]  log_coeffs
  const int tid  = threadIdx.x;
  const int lane = tid & 31;
  const int wid  = tid >> 5;
  const int r    = lane & 15;
  const int hf   = lane >> 4;

  const int waveM = wid & 3;          // 4 waves along M
  const int waveN = wid >> 2;         // 2 waves along paired-N
  const int m0 = blockIdx.y * 128 + waveM * 32;
  const int n0 = blockIdx.x * 64  + waveN * 32;   // gate column base (hidden = +512)

  v8f accG[2][2] = {};
  v8f accH[2][2] = {};

  for (int k = 0; k < KDIM; k += 32) {
    // prefetch next K-slab of A rows (hits L2; x/h-bf16 is fully L2 resident)
    if (k + 32 < KDIM)
      __builtin_prefetch(X + (size_t)(m0 + r) * KDIM + k + 32, 0, 1);

    v16bf a0 = load_frag(X, m0,         k, r, hf);
    v16bf a1 = load_frag(X, m0 + 16,    k, r, hf);
    v16bf g0 = load_frag(W, n0,         k, r, hf);
    v16bf g1 = load_frag(W, n0 + 16,    k, r, hf);
    v16bf h0 = load_frag(W, H_ + n0,      k, r, hf);
    v16bf h1 = load_frag(W, H_ + n0 + 16, k, r, hf);

    accG[0][0] = __builtin_amdgcn_wmma_f32_16x16x32_bf16(false, a0, false, g0, (short)0, accG[0][0], false, false);
    accG[0][1] = __builtin_amdgcn_wmma_f32_16x16x32_bf16(false, a0, false, g1, (short)0, accG[0][1], false, false);
    accG[1][0] = __builtin_amdgcn_wmma_f32_16x16x32_bf16(false, a1, false, g0, (short)0, accG[1][0], false, false);
    accG[1][1] = __builtin_amdgcn_wmma_f32_16x16x32_bf16(false, a1, false, g1, (short)0, accG[1][1], false, false);
    accH[0][0] = __builtin_amdgcn_wmma_f32_16x16x32_bf16(false, a0, false, h0, (short)0, accH[0][0], false, false);
    accH[0][1] = __builtin_amdgcn_wmma_f32_16x16x32_bf16(false, a0, false, h1, (short)0, accH[0][1], false, false);
    accH[1][0] = __builtin_amdgcn_wmma_f32_16x16x32_bf16(false, a1, false, h0, (short)0, accH[1][0], false, false);
    accH[1][1] = __builtin_amdgcn_wmma_f32_16x16x32_bf16(false, a1, false, h1, (short)0, accH[1][1], false, false);
  }

  // Epilogue: same lane holds matching gate/hidden elements -> transform in regs.
  // C/D layout: lane L: N = L%16; VGPR idx: M = idx + (L<16 ? 0 : 8).
#pragma unroll
  for (int mi = 0; mi < 2; ++mi) {
#pragma unroll
    for (int ni = 0; ni < 2; ++ni) {
      const int n  = n0 + ni * 16 + r;
      const float bg = bias[n];
      const float bh = bias[H_ + n];
#pragma unroll
      for (int idx = 0; idx < 8; ++idx) {
        const int m  = m0 + mi * 16 + idx + hf * 8;
        const float g  = accG[mi][ni][idx] + bg;
        const float hd = accH[mi][ni][idx] + bh;
        const size_t o = (size_t)m * H_ + n;
        cOut[o] = -softplusf_(g);                    // log_coeffs
        aOut[o] = -softplusf_(-g) + log_g_(hd);      // log_z + log_tilde_h
      }
    }
  }
}

// ---------------- chunked parallel log-space scan ----------------
// Step: L_t = logaddexp(c_t + L_{t-1}, v_t).  Chunk operator: (Ctot, V).

__global__ void k_scan_phase1(const float* __restrict__ aBuf, const float* __restrict__ cBuf,
                              float* __restrict__ chC, float* __restrict__ chV) {
  const int j = blockIdx.x * blockDim.x + threadIdx.x;  // channel
  const int chunk = blockIdx.y;
  const int b = blockIdx.z;
  size_t base = ((size_t)b * S_ + (size_t)chunk * CL) * H_ + j;
  float Ct = 0.0f, L = -INFINITY;
  for (int i = 0; i < CL; ++i) {
    const size_t o = base + (size_t)i * H_;
    const float c = cBuf[o];
    Ct += c;
    L = logaddexpf_(c + L, aBuf[o]);
  }
  const size_t si = ((size_t)(b * H_ + j)) * NC + chunk;
  chC[si] = Ct;
  chV[si] = L;
}

__global__ void k_scan_phase2(const float* __restrict__ chC, const float* __restrict__ chV,
                              float* __restrict__ carry, float* __restrict__ nh) {
  const int seq = blockIdx.x * blockDim.x + threadIdx.x;  // b*H + j, 4096 total
  const int b = seq >> 9, j = seq & (H_ - 1);
  const size_t si = (size_t)seq * NC;
  float L = logf(0.5f);  // log(h0)
  for (int c = 0; c < NC; ++c) {
    carry[si + c] = L;
    L = logaddexpf_(chC[si + c] + L, chV[si + c]);
  }
  nh[(size_t)b * H_ + j] = expf(L);  // next_hidden (== h at t = S-1)
}

__global__ void k_scan_phase3(const float* __restrict__ aBuf, const float* __restrict__ cBuf,
                              const float* __restrict__ carry,
                              float* __restrict__ outF, unsigned short* __restrict__ outB) {
  const int j = blockIdx.x * blockDim.x + threadIdx.x;
  const int chunk = blockIdx.y;
  const int b = blockIdx.z;
  size_t base = ((size_t)b * S_ + (size_t)chunk * CL) * H_ + j;
  float L = carry[((size_t)(b * H_ + j)) * NC + chunk];
  for (int i = 0; i < CL; ++i) {
    const size_t o = base + (size_t)i * H_;
    L = logaddexpf_(cBuf[o] + L, aBuf[o]);
    const float h = expf(L);
    if (outF) outF[o] = h;
    else      outB[o] = f2bf(h);   // feeds next layer's WMMA directly
  }
}

// ---------------- launch ----------------

extern "C" void kernel_launch(void* const* d_in, const int* in_sizes, int n_in,
                              void* d_out, int out_size, void* d_ws, size_t ws_size,
                              hipStream_t stream) {
  (void)in_sizes; (void)n_in; (void)out_size; (void)ws_size;
  const float* x  = (const float*)d_in[0];
  const float* w0 = (const float*)d_in[1];
  const float* b0 = (const float*)d_in[2];
  const float* w1 = (const float*)d_in[3];
  const float* b1 = (const float*)d_in[4];
  float* out = (float*)d_out;

  char* ws = (char*)d_ws;
  size_t off = 0;
  auto wsAlloc = [&](size_t bytes) -> void* {
    void* p = ws + off;
    off += (bytes + 255) & ~(size_t)255;
    return p;
  };
  unsigned short* xbf  = (unsigned short*)wsAlloc((size_t)M_ * KDIM * 2);  // also reused for layer-0 h (bf16)
  unsigned short* w0bf = (unsigned short*)wsAlloc((size_t)1024 * 512 * 2);
  unsigned short* w1bf = (unsigned short*)wsAlloc((size_t)1024 * 512 * 2);
  float* aBuf  = (float*)wsAlloc((size_t)M_ * H_ * 4);
  float* cBuf  = (float*)wsAlloc((size_t)M_ * H_ * 4);
  float* chC   = (float*)wsAlloc((size_t)B_ * H_ * NC * 4);
  float* chV   = (float*)wsAlloc((size_t)B_ * H_ * NC * 4);
  float* carry = (float*)wsAlloc((size_t)B_ * H_ * NC * 4);

  float* nh = out + (size_t)M_ * H_;   // next_hidden tail: (2, B, 1, H)

  // precision staging
  k_f32_to_bf16<<<(M_ * KDIM + 255) / 256, 256, 0, stream>>>(x, xbf, M_ * KDIM);
  k_f32_to_bf16<<<(1024 * 512 + 255) / 256, 256, 0, stream>>>(w0, w0bf, 1024 * 512);
  k_f32_to_bf16<<<(1024 * 512 + 255) / 256, 256, 0, stream>>>(w1, w1bf, 1024 * 512);

  const dim3 gGemm(H_ / 64, M_ / 128);         // (8, 256)
  const dim3 gScan(H_ / 256, NC, B_);          // (2, 32, 8)

  // ---- layer 0 ----
  k_gemm_fused<<<gGemm, 256, 0, stream>>>(xbf, w0bf, b0, aBuf, cBuf);
  k_scan_phase1<<<gScan, 256, 0, stream>>>(aBuf, cBuf, chC, chV);
  k_scan_phase2<<<(B_ * H_) / 256, 256, 0, stream>>>(chC, chV, carry, nh);
  k_scan_phase3<<<gScan, 256, 0, stream>>>(aBuf, cBuf, carry, nullptr, xbf);  // h0 -> bf16 (reuse xbf)

  // ---- layer 1 ----
  k_gemm_fused<<<gGemm, 256, 0, stream>>>(xbf, w1bf, b1, aBuf, cBuf);
  k_scan_phase1<<<gScan, 256, 0, stream>>>(aBuf, cBuf, chC, chV);
  k_scan_phase2<<<(B_ * H_) / 256, 256, 0, stream>>>(chC, chV, carry, nh + (size_t)B_ * H_);
  k_scan_phase3<<<gScan, 256, 0, stream>>>(aBuf, cBuf, carry, out, nullptr);
}
